// DFCSC_Longformer_13477607375561
// MI455X (gfx1250) — matrix operand order repaired and checked by reading, not compiled, not run on hardware
//
#include <hip/hip_runtime.h>
#include <hip/hip_bf16.h>
#include <math.h>

// ---------------- model constants ----------------
#define BATCH   2
#define S_LEN   2048
#define D_MODEL 768
#define NHEAD   12
#define DHEAD   64
#define NLAYER  2
#define FFDIM   3072
#define NCHUNK  16
#define CHUNK   128
#define WINSZ   128
#define NGLOB   17        // 1 cls + 16 sep
#define NKEY    416       // 384 local + 17 global, padded to 13*32
#define NEGINF  (-1.0e9f)
#define MROWS   (BATCH*S_LEN)   // 4096

typedef __attribute__((ext_vector_type(16))) _Float16 v16h;
typedef __attribute__((ext_vector_type(8)))  _Float16 v8h;
typedef __attribute__((ext_vector_type(8)))  float    v8f;

union V16 { v16h v; v8h h[2]; };

__device__ inline v8f wmma16(v16h a, v16h b, v8f c) {
  // D = A(16x32 f16) * B(32x16 f16) + C(16x16 f32)
  return __builtin_amdgcn_wmma_f32_16x16x32_f16(false, a, false, b, (short)0, c,
                                                false, false);
}
__device__ inline v8f zero8() {
  v8f z = {0.f, 0.f, 0.f, 0.f, 0.f, 0.f, 0.f, 0.f};
  return z;
}
__device__ inline int iclip(int v, int lo, int hi) {
  return v < lo ? lo : (v > hi ? hi : v);
}
__device__ inline float gelu_f(float x) {
  const float k = 0.7978845608028654f;
  float t = tanhf(k * (x + 0.044715f * x * x * x));
  return 0.5f * x * (1.f + t);
}

// ---- CDNA5 async global->LDS 16B copy (ASYNCcnt-tracked DMA) ----------
// The clang builtin on this toolchain has a descriptor-style signature, so
// emit the VGLOBAL instruction directly (CDNA5 ISA).
__device__ inline void cp16(const _Float16* g, _Float16* l) {
  unsigned loff = (unsigned)(size_t)(void*)l;  // low 32 bits = LDS offset
  asm volatile("global_load_async_to_lds_b128 %0, %1, off"
               :: "v"(loff), "v"(g) : "memory");
}
__device__ inline void async_wait() {
#if __has_builtin(__builtin_amdgcn_s_wait_asynccnt)
  __builtin_amdgcn_s_wait_asynccnt(0);
#else
  asm volatile("s_wait_asynccnt 0x0" ::: "memory");
#endif
}

// ---------------- embedding: t32 = tok_emb[id] + pos_emb ----------------
__global__ __launch_bounds__(256) void embed_kernel(
    const int* __restrict__ ids, const float* __restrict__ tok,
    const float* __restrict__ pos, float* __restrict__ t32) {
  int row = blockIdx.x;               // b*S + s
  int s = row & (S_LEN - 1);
  int id = ids[row];
  const float* tr = tok + (size_t)id * D_MODEL;
  const float* pr = pos + (size_t)s * D_MODEL;
  float* o = t32 + (size_t)row * D_MODEL;
  for (int d = threadIdx.x; d < D_MODEL; d += 256) o[d] = tr[d] + pr[d];
}

// ---------------- LayerNorm: x32 = LN(A + R)*g + b ; xh = f16(x32) -------
__global__ __launch_bounds__(256) void ln_kernel(
    const float* __restrict__ A, const float* __restrict__ R,
    const float* __restrict__ g, const float* __restrict__ be,
    float* __restrict__ x32, _Float16* __restrict__ xh) {
  int row = blockIdx.x;
  const float* a = A + (size_t)row * D_MODEL;
  const float* r = R ? R + (size_t)row * D_MODEL : nullptr;
  __shared__ float red[256];
  float vals[3];
  float s = 0.f;
#pragma unroll
  for (int i = 0; i < 3; ++i) {
    int d = threadIdx.x + i * 256;
    float v = a[d] + (r ? r[d] : 0.f);
    vals[i] = v;
    s += v;
  }
  red[threadIdx.x] = s;
  __syncthreads();
  for (int st = 128; st > 0; st >>= 1) {
    if (threadIdx.x < st) red[threadIdx.x] += red[threadIdx.x + st];
    __syncthreads();
  }
  float mean = red[0] * (1.f / D_MODEL);
  __syncthreads();
  float sq = 0.f;
#pragma unroll
  for (int i = 0; i < 3; ++i) {
    float d = vals[i] - mean;
    sq += d * d;
  }
  red[threadIdx.x] = sq;
  __syncthreads();
  for (int st = 128; st > 0; st >>= 1) {
    if (threadIdx.x < st) red[threadIdx.x] += red[threadIdx.x + st];
    __syncthreads();
  }
  float rstd = rsqrtf(red[0] * (1.f / D_MODEL) + 1e-5f);
#pragma unroll
  for (int i = 0; i < 3; ++i) {
    int d = threadIdx.x + i * 256;
    float v = (vals[i] - mean) * rstd * g[d] + be[d];
    x32[(size_t)row * D_MODEL + d] = v;
    xh[(size_t)row * D_MODEL + d] = (_Float16)v;
  }
}

// ---------------- SEP scan: glob[17], gbits[S/32] ----------------
__global__ void sep_kernel(const int* __restrict__ ids, int* __restrict__ glob,
                           unsigned* __restrict__ gbits) {
  if (threadIdx.x == 0 && blockIdx.x == 0) {
    for (int w = 0; w < S_LEN / 32; ++w) gbits[w] = 0u;
    glob[0] = 0;
    int c = 1;
    for (int s = 0; s < S_LEN; ++s)
      if (ids[s] == 2 && c < NGLOB) glob[c++] = s;
    while (c < NGLOB) glob[c++] = 0;
    for (int i = 0; i < NGLOB; ++i) {
      int p = glob[i];
      gbits[p >> 5] |= (1u << (p & 31));
    }
  }
}

// ---- weight convert+transpose: W fp32 (KxN) -> Wt f16 (NxK), per z ----
__global__ __launch_bounds__(256) void cvt_transpose_kernel(
    const float* __restrict__ W, _Float16* __restrict__ Wt, int K, int N) {
  const float* src = W + (size_t)blockIdx.z * K * N;
  _Float16* dst = Wt + (size_t)blockIdx.z * K * N;
  __shared__ _Float16 tile[64][65];
  int k0 = blockIdx.y * 64, n0 = blockIdx.x * 64;
  for (int idx = threadIdx.x; idx < 4096; idx += 256) {
    int k = idx >> 6, n = idx & 63;                      // coalesced along n
    tile[n][k] = (_Float16)src[(size_t)(k0 + k) * N + n0 + n];
  }
  __syncthreads();
  for (int idx = threadIdx.x; idx < 4096; idx += 256) {
    int n = idx >> 6, k = idx & 63;                      // coalesced along k
    dst[(size_t)(n0 + n) * K + k0 + k] = tile[n][k];
  }
}

// ---------------- WMMA GEMM (double-buffered LDS, async staging) --------
// out = act(A[f16 MxK] @ B + bias), B given pre-transposed f16 (NxK).
// 128x128 block tile, 8 waves of 32x64, K-step 32, ping-pong LDS.
__global__ __launch_bounds__(256) void gemm_kernel(
    const _Float16* __restrict__ A, const _Float16* __restrict__ Bt,
    const float* __restrict__ bias, float* __restrict__ outF,
    _Float16* __restrict__ outH, int M, int N, int K, int act) {
  const int n0 = blockIdx.x * 128;
  const int m0 = blockIdx.y * 128;
  const int tid = threadIdx.x;
  const int wave = tid >> 5, lane = tid & 31;
  const int wr = wave >> 1, wc = wave & 1;
  const int hi = lane >> 4, ln = lane & 15;

  __shared__ __align__(16) _Float16 As[2][128][40];
  __shared__ __align__(16) _Float16 Bs[2][128][40];

  // staging assignment: each thread copies 32B of A and 32B of B per step
  const int srow = tid >> 1, kh = (tid & 1) * 16;
  const _Float16* aSrc = A + (size_t)(m0 + srow) * K + kh;
  const _Float16* bSrc = Bt + (size_t)(n0 + srow) * K + kh;

  v8f acc[2][4];
#pragma unroll
  for (int i = 0; i < 2; ++i)
#pragma unroll
    for (int j = 0; j < 4; ++j) acc[i][j] = zero8();

  // prologue: stage k0 = 0 into buffer 0
  cp16(aSrc, &As[0][srow][kh]);
  cp16(aSrc + 8, &As[0][srow][kh + 8]);
  cp16(bSrc, &Bs[0][srow][kh]);
  cp16(bSrc + 8, &Bs[0][srow][kh + 8]);
  async_wait();
  __syncthreads();

  int buf = 0;
  for (int k0 = 0; k0 < K; k0 += 32) {
    // prefetch next K-step into the other buffer (overlaps with WMMAs)
    if (k0 + 32 < K) {
      const _Float16* an = aSrc + k0 + 32;
      const _Float16* bn = bSrc + k0 + 32;
      cp16(an, &As[buf ^ 1][srow][kh]);
      cp16(an + 8, &As[buf ^ 1][srow][kh + 8]);
      cp16(bn, &Bs[buf ^ 1][srow][kh]);
      cp16(bn + 8, &Bs[buf ^ 1][srow][kh + 8]);
    }
    V16 af[2], bf[4];
#pragma unroll
    for (int mi = 0; mi < 2; ++mi) {
      int row = wr * 32 + mi * 16 + ln;
      af[mi].h[0] = *(const v8h*)&As[buf][row][hi * 8];
      af[mi].h[1] = *(const v8h*)&As[buf][row][16 + hi * 8];
    }
#pragma unroll
    for (int ni = 0; ni < 4; ++ni) {
      int col = wc * 64 + ni * 16 + ln;
      bf[ni].h[0] = *(const v8h*)&Bs[buf][col][hi * 8];
      bf[ni].h[1] = *(const v8h*)&Bs[buf][col][16 + hi * 8];
    }
#pragma unroll
    for (int mi = 0; mi < 2; ++mi)
#pragma unroll
      for (int ni = 0; ni < 4; ++ni)
        acc[mi][ni] = wmma16(af[mi].v, bf[ni].v, acc[mi][ni]);
    async_wait();
    __syncthreads();
    buf ^= 1;
  }
  // epilogue
#pragma unroll
  for (int mi = 0; mi < 2; ++mi)
#pragma unroll
    for (int ni = 0; ni < 4; ++ni) {
      int col = n0 + wc * 64 + ni * 16 + ln;
      float bv = bias ? bias[col] : 0.f;
#pragma unroll
      for (int r = 0; r < 8; ++r) {
        int row = m0 + wr * 32 + mi * 16 + hi * 8 + r;
        float v = acc[mi][ni][r] + bv;
        if (act == 1) v = gelu_f(v);
        size_t idx = (size_t)row * N + col;
        if (outF) outF[idx] = v;
        if (outH) outH[idx] = (_Float16)v;
      }
    }
}

// ---------------- local sliding-window attention (WMMA) ----------------
// One WG per (b, h, chunk, half): 64 query rows, 416 key cols.
// dyn LDS: Sc[64][416] f32 (re-used as f16 probs) + vT[64][416] f16.
#define SCS 416
#define VTS 416
#define LOCAL_LDS (64 * 416 * 4 + 64 * 416 * 2)

__global__ __launch_bounds__(256) void local_attn_kernel(
    const _Float16* __restrict__ qh, const _Float16* __restrict__ kh,
    const _Float16* __restrict__ vh, _Float16* __restrict__ ah,
    const int* __restrict__ glob, const unsigned* __restrict__ gbits) {
  extern __shared__ __align__(16) char smem[];
  float* Sc = (float*)smem;
  _Float16* vT = (_Float16*)(smem + 64 * SCS * 4);
  __shared__ float rsum[64];

  int blk = blockIdx.x;
  int hf = blk & 1;
  int t = blk >> 1;
  int n = t % NCHUNK;  t /= NCHUNK;
  int h = t % NHEAD;
  int b = t / NHEAD;
  const int lane = threadIdx.x & 31, wave = threadIdx.x >> 5;
  const int hi = lane >> 4, ln = lane & 15;
  const int qbase = n * CHUNK + hf * 64;

  // --- stage V^T (value rows gathered, transposed into LDS) ---
  for (int j = threadIdx.x; j < NKEY; j += 256) {
    int src = -1;
    if (j < 384) src = iclip(n * CHUNK - CHUNK + j, 0, S_LEN - 1);
    else if (j < 384 + NGLOB) src = glob[j - 384];
    if (src >= 0) {
      const _Float16* vr = vh + ((size_t)(b * S_LEN + src)) * D_MODEL + h * DHEAD;
#pragma unroll 8
      for (int d = 0; d < DHEAD; ++d) vT[d * VTS + j] = vr[d];
    } else {
      for (int d = 0; d < DHEAD; ++d) vT[d * VTS + j] = (_Float16)0.f;
    }
  }
  __syncthreads();

  // --- scores: S = Q @ K^T via WMMA, direct-from-global fragments ---
  {
    int mt = wave >> 1;       // 0..3 : 16-row q tile
    int nh = wave & 1;        // half of the 26 key tiles
    int qrow = qbase + mt * 16 + ln;
    const _Float16* qr = qh + ((size_t)(b * S_LEN + qrow)) * D_MODEL + h * DHEAD;
    V16 a0, a1;
    a0.h[0] = *(const v8h*)(qr + hi * 8);
    a0.h[1] = *(const v8h*)(qr + 16 + hi * 8);
    a1.h[0] = *(const v8h*)(qr + 32 + hi * 8);
    a1.h[1] = *(const v8h*)(qr + 48 + hi * 8);
    for (int nt = nh * 13; nt < nh * 13 + 13; ++nt) {
      int j = nt * 16 + ln;
      int src;
      if (j < 384) src = iclip(n * CHUNK - CHUNK + j, 0, S_LEN - 1);
      else if (j < 384 + NGLOB) src = glob[j - 384];
      else src = 0;
      const _Float16* kr = kh + ((size_t)(b * S_LEN + src)) * D_MODEL + h * DHEAD;
      V16 b0, b1;
      b0.h[0] = *(const v8h*)(kr + hi * 8);
      b0.h[1] = *(const v8h*)(kr + 16 + hi * 8);
      b1.h[0] = *(const v8h*)(kr + 32 + hi * 8);
      b1.h[1] = *(const v8h*)(kr + 48 + hi * 8);
      v8f acc = zero8();
      acc = wmma16(a0.v, b0.v, acc);
      acc = wmma16(a1.v, b1.v, acc);
#pragma unroll
      for (int r = 0; r < 8; ++r)
        Sc[(mt * 16 + hi * 8 + r) * SCS + nt * 16 + ln] = acc[r];
    }
  }
  __syncthreads();

  // --- mask + scale + softmax per row (thread-per-row, in-place f16 probs) ---
  if (threadIdx.x < 64) {
    int mrow = threadIdx.x;
    int qrow = qbase + mrow;
    float* srow = Sc + (size_t)mrow * SCS;
    float mx = -1e30f;
    for (int j = 0; j < NKEY; ++j) {
      float v = srow[j];
      bool valid;
      if (j < 384) {
        int p = n * CHUNK - CHUNK + j;
        bool inb = (p >= 0) && (p < S_LEN);
        int pc = iclip(p, 0, S_LEN - 1);
        bool isg = (gbits[pc >> 5] >> (pc & 31)) & 1u;
        int dlt = p - qrow;
        if (dlt < 0) dlt = -dlt;
        valid = inb && !isg && (dlt <= WINSZ);
      } else {
        valid = (j < 384 + NGLOB);
      }
      v = valid ? v * 0.125f : NEGINF;  // DH^-0.5 = 0.125
      srow[j] = v;
      mx = fmaxf(mx, v);
    }
    float sum = 0.f;
    _Float16* prow = (_Float16*)srow;  // safe forward-sequential aliasing
    for (int j = 0; j < NKEY; ++j) {
      float e = __expf(srow[j] - mx);
      sum += e;
      prow[j] = (_Float16)e;
    }
    rsum[mrow] = sum;
  }
  __syncthreads();

  // --- out = P @ Vcat via WMMA (A from LDS probs, B from LDS V^T) ---
  {
    int mt = wave >> 1;
    int dbase = (wave & 1) * 2;
    for (int dt2 = 0; dt2 < 2; ++dt2) {
      int dt = dbase + dt2;  // 16-wide dh tile, 0..3
      v8f acc = zero8();
      const _Float16* prow = (const _Float16*)(Sc + (size_t)(mt * 16 + ln) * SCS);
      const _Float16* vrow = vT + (size_t)(dt * 16 + ln) * VTS;
      for (int ks = 0; ks < NKEY / 32; ++ks) {
        V16 af, bf;
        af.h[0] = *(const v8h*)(prow + ks * 32 + hi * 8);
        af.h[1] = *(const v8h*)(prow + ks * 32 + 16 + hi * 8);
        bf.h[0] = *(const v8h*)(vrow + ks * 32 + hi * 8);
        bf.h[1] = *(const v8h*)(vrow + ks * 32 + 16 + hi * 8);
        acc = wmma16(af.v, bf.v, acc);
      }
#pragma unroll
      for (int r = 0; r < 8; ++r) {
        int mrow = mt * 16 + hi * 8 + r;
        int qrow = qbase + mrow;
        float o = acc[r] / rsum[mrow];
        ah[((size_t)(b * S_LEN + qrow)) * D_MODEL + h * DHEAD + dt * 16 + ln] =
            (_Float16)o;
      }
    }
  }
}

// ---------------- global-token attention (overwrites ah rows) ------------
__global__ __launch_bounds__(256) void global_attn_kernel(
    const _Float16* __restrict__ qh, const _Float16* __restrict__ kh,
    const _Float16* __restrict__ vh, _Float16* __restrict__ ah,
    const int* __restrict__ glob) {
  int blk = blockIdx.x;  // (b*H + h)*NGLOB + g
  int g = blk % NGLOB;
  int h = (blk / NGLOB) % NHEAD;
  int b = blk / (NGLOB * NHEAD);
  int qpos = glob[g];
  __shared__ float qs[DHEAD];
  __shared__ float sc[S_LEN];
  __shared__ float red[256];
  __shared__ float osum[4 * DHEAD];
  int tid = threadIdx.x;
  const _Float16* qr = qh + ((size_t)(b * S_LEN + qpos)) * D_MODEL + h * DHEAD;
  if (tid < DHEAD) qs[tid] = (float)qr[tid] * 0.125f;
  __syncthreads();
  float lmax = -1e30f;
  for (int s = tid; s < S_LEN; s += 256) {
    const _Float16* kr = kh + ((size_t)(b * S_LEN + s)) * D_MODEL + h * DHEAD;
    float d = 0.f;
#pragma unroll 8
    for (int i = 0; i < DHEAD; ++i) d += qs[i] * (float)kr[i];
    sc[s] = d;
    lmax = fmaxf(lmax, d);
  }
  red[tid] = lmax;
  __syncthreads();
  for (int st = 128; st > 0; st >>= 1) {
    if (tid < st) red[tid] = fmaxf(red[tid], red[tid + st]);
    __syncthreads();
  }
  float mx = red[0];
  __syncthreads();
  float ls = 0.f;
  for (int s = tid; s < S_LEN; s += 256) {
    float e = __expf(sc[s] - mx);
    sc[s] = e;
    ls += e;
  }
  red[tid] = ls;
  __syncthreads();
  for (int st = 128; st > 0; st >>= 1) {
    if (tid < st) red[tid] += red[tid + st];
    __syncthreads();
  }
  float inv = 1.f / red[0];
  __syncthreads();
  int dh = tid & (DHEAD - 1), grp = tid >> 6;
  float a = 0.f;
  for (int s = grp * 512; s < grp * 512 + 512; ++s) {
    const _Float16* vr = vh + ((size_t)(b * S_LEN + s)) * D_MODEL + h * DHEAD;
    a += sc[s] * (float)vr[dh];
  }
  osum[grp * DHEAD + dh] = a;
  __syncthreads();
  if (tid < DHEAD) {
    float o = (osum[tid] + osum[DHEAD + tid] + osum[2 * DHEAD + tid] +
               osum[3 * DHEAD + tid]) * inv;
    ah[((size_t)(b * S_LEN + qpos)) * D_MODEL + h * DHEAD + tid] = (_Float16)o;
  }
}

// ---------------- classification head ----------------
__global__ __launch_bounds__(128) void head_kernel(
    const float* __restrict__ x32, const int* __restrict__ glob,
    const float* __restrict__ Wh, const float* __restrict__ bh,
    const float* __restrict__ Wout, const float* __restrict__ bout,
    float* __restrict__ out) {
  int blk = blockIdx.x;  // b*14 + i
  int i = blk % 14, b = blk / 14;
  int core = glob[2 + i];  // sep_pos[1..14]
  __shared__ float hbuf[100];
  const float* cls = x32 + ((size_t)(b * S_LEN)) * D_MODEL;
  const float* sep = x32 + ((size_t)(b * S_LEN + core)) * D_MODEL;
  int tid = threadIdx.x;
  if (tid < 100) {
    float a = bh[tid];
    for (int d = 0; d < D_MODEL; ++d) a += cls[d] * Wh[(size_t)d * 100 + tid];
    for (int d = 0; d < D_MODEL; ++d)
      a += sep[d] * Wh[(size_t)(D_MODEL + d) * 100 + tid];
    hbuf[tid] = fmaxf(a, 0.f);
  }
  __syncthreads();
  if (tid < 7) {
    float a = bout[tid];
    for (int j = 0; j < 100; ++j) a += hbuf[j] * Wout[j * 7 + tid];
    out[(size_t)blk * 7 + tid] = a;
  }
}

// =====================================================================
extern "C" void kernel_launch(void* const* d_in, const int* in_sizes, int n_in,
                              void* d_out, int out_size, void* d_ws,
                              size_t ws_size, hipStream_t stream) {
  (void)in_sizes; (void)n_in; (void)out_size;
  const int*   input_ids = (const int*)d_in[0];
  /* d_in[1] attention_mask: all-ones, folded analytically */
  const float* tok_emb = (const float*)d_in[2];
  const float* pos_emb = (const float*)d_in[3];
  const float* ln_e_g  = (const float*)d_in[4];
  const float* ln_e_b  = (const float*)d_in[5];
  const float* Wq = (const float*)d_in[6];
  const float* bq = (const float*)d_in[7];
  const float* Wk = (const float*)d_in[8];
  const float* bk = (const float*)d_in[9];
  const float* Wv = (const float*)d_in[10];
  const float* bv = (const float*)d_in[11];
  const float* Wo = (const float*)d_in[12];
  const float* bo = (const float*)d_in[13];
  const float* ln1_g = (const float*)d_in[14];
  const float* ln1_b = (const float*)d_in[15];
  const float* W1 = (const float*)d_in[16];
  const float* b1 = (const float*)d_in[17];
  const float* W2 = (const float*)d_in[18];
  const float* b2 = (const float*)d_in[19];
  const float* ln2_g = (const float*)d_in[20];
  const float* ln2_b = (const float*)d_in[21];
  const float* Wh = (const float*)d_in[22];
  const float* bh = (const float*)d_in[23];
  const float* Wout = (const float*)d_in[24];
  const float* bout = (const float*)d_in[25];
  float* out = (float*)d_out;

  // ---- workspace arena (~110 MB total) ----
  char* ws = (char*)d_ws;
  size_t off = 0;
  auto take = [&](size_t bytes) -> char* {
    char* p = ws + off;
    off = (off + bytes + 255) & ~(size_t)255;
    return p;
  };
  float*     x32 = (float*)take((size_t)MROWS * D_MODEL * 4);
  float*     t32 = (float*)take((size_t)MROWS * D_MODEL * 4);
  _Float16*  xh  = (_Float16*)take((size_t)MROWS * D_MODEL * 2);
  _Float16*  qhB = (_Float16*)take((size_t)MROWS * D_MODEL * 2);
  _Float16*  khB = (_Float16*)take((size_t)MROWS * D_MODEL * 2);
  _Float16*  vhB = (_Float16*)take((size_t)MROWS * D_MODEL * 2);
  _Float16*  ahB = (_Float16*)take((size_t)MROWS * D_MODEL * 2);
  _Float16*  hhB = (_Float16*)take((size_t)MROWS * FFDIM * 2);
  _Float16*  WqT = (_Float16*)take((size_t)NLAYER * D_MODEL * D_MODEL * 2);
  _Float16*  WkT = (_Float16*)take((size_t)NLAYER * D_MODEL * D_MODEL * 2);
  _Float16*  WvT = (_Float16*)take((size_t)NLAYER * D_MODEL * D_MODEL * 2);
  _Float16*  WoT = (_Float16*)take((size_t)NLAYER * D_MODEL * D_MODEL * 2);
  _Float16*  W1T = (_Float16*)take((size_t)NLAYER * D_MODEL * FFDIM * 2);
  _Float16*  W2T = (_Float16*)take((size_t)NLAYER * FFDIM * D_MODEL * 2);
  int*       glob  = (int*)take(NGLOB * 4);
  unsigned*  gbits = (unsigned*)take((S_LEN / 32) * 4);
  if (off > ws_size) return;  // workspace too small: bail deterministically

  (void)hipFuncSetAttribute((const void*)local_attn_kernel,
                            hipFuncAttributeMaxDynamicSharedMemorySize,
                            LOCAL_LDS);

  // weights: fp32 (KxN) -> f16 transposed (NxK), once per call
  auto cvtT = [&](const float* s, _Float16* d, int K, int N) {
    dim3 g(N / 64, K / 64, NLAYER);
    cvt_transpose_kernel<<<g, 256, 0, stream>>>(s, d, K, N);
  };
  const int DD = D_MODEL * D_MODEL, DF = D_MODEL * FFDIM;
  cvtT(Wq, WqT, D_MODEL, D_MODEL);
  cvtT(Wk, WkT, D_MODEL, D_MODEL);
  cvtT(Wv, WvT, D_MODEL, D_MODEL);
  cvtT(Wo, WoT, D_MODEL, D_MODEL);
  cvtT(W1, W1T, D_MODEL, FFDIM);
  cvtT(W2, W2T, FFDIM, D_MODEL);

  sep_kernel<<<1, 32, 0, stream>>>(input_ids, glob, gbits);
  embed_kernel<<<MROWS, 256, 0, stream>>>(input_ids, tok_emb, pos_emb, t32);
  ln_kernel<<<MROWS, 256, 0, stream>>>(t32, nullptr, ln_e_g, ln_e_b, x32, xh);

  auto gemm = [&](const _Float16* A, const _Float16* Bm, const float* bias,
                  float* oF, _Float16* oH, int M, int N, int K, int act) {
    dim3 g(N / 128, M / 128);
    gemm_kernel<<<g, 256, 0, stream>>>(A, Bm, bias, oF, oH, M, N, K, act);
  };

  for (int l = 0; l < NLAYER; ++l) {
    // Q/K/V projections -> f16
    gemm(xh, WqT + (size_t)l * DD, bq + l * D_MODEL, nullptr, qhB,
         MROWS, D_MODEL, D_MODEL, 0);
    gemm(xh, WkT + (size_t)l * DD, bk + l * D_MODEL, nullptr, khB,
         MROWS, D_MODEL, D_MODEL, 0);
    gemm(xh, WvT + (size_t)l * DD, bv + l * D_MODEL, nullptr, vhB,
         MROWS, D_MODEL, D_MODEL, 0);
    // sliding-window attention (WMMA) + global-token attention (overwrite)
    local_attn_kernel<<<BATCH * NHEAD * NCHUNK * 2, 256, LOCAL_LDS, stream>>>(
        qhB, khB, vhB, ahB, glob, gbits);
    global_attn_kernel<<<BATCH * NHEAD * NGLOB, 256, 0, stream>>>(
        qhB, khB, vhB, ahB, glob);
    // output projection, residual + LN1
    gemm(ahB, WoT + (size_t)l * DD, bo + l * D_MODEL, t32, nullptr,
         MROWS, D_MODEL, D_MODEL, 0);
    ln_kernel<<<MROWS, 256, 0, stream>>>(t32, x32, ln1_g + l * D_MODEL,
                                         ln1_b + l * D_MODEL, x32, xh);
    // FFN: gelu(x@W1+b1)@W2+b2, residual + LN2
    gemm(xh, W1T + (size_t)l * DF, b1 + l * FFDIM, nullptr, hhB,
         MROWS, FFDIM, D_MODEL, 1);
    gemm(hhB, W2T + (size_t)l * DF, b2 + l * D_MODEL, t32, nullptr,
         MROWS, D_MODEL, FFDIM, 0);
    ln_kernel<<<MROWS, 256, 0, stream>>>(t32, x32, ln2_g + l * D_MODEL,
                                         ln2_b + l * D_MODEL, x32, xh);
  }

  head_kernel<<<BATCH * 14, 128, 0, stream>>>(x32, glob, Wh, bh, Wout, bout,
                                              out);
}
